// CPGA_82892868813324
// MI455X (gfx1250) — compile-verified, matrix operand order checked by reading, not run on hardware
//
#include <hip/hip_runtime.h>
#include <hip/hip_bf16.h>
#include <math.h>

// ---------------- problem constants ----------------
constexpr int kB  = 4;
constexpr int kC  = 256;
constexpr int kH  = 128;
constexpr int kW  = 128;
constexpr int kHW = kH * kW;          // 16384 = 2^14
constexpr int kM  = kB * kHW;         // 65536 pixel rows
constexpr int kNC = 19;
constexpr int kNH = 8;
constexpr int kHD = kC / kNH;         // 32
constexpr float kSCALE = 0.17677669529663687f; // 32^-0.5
constexpr float kMOM = 0.1f;
constexpr int kC4 = 4 * kC;           // 1024

typedef __attribute__((ext_vector_type(16))) __bf16 v16bf;
typedef __attribute__((ext_vector_type(8)))  float  v8f;

__device__ __forceinline__ __bf16 f2bf(float f) {
  union { float f; unsigned u; } c; c.f = f;
  unsigned r = c.u + 0x7FFFu + ((c.u >> 16) & 1u);
  unsigned short h = (unsigned short)(r >> 16);
  __bf16 o; __builtin_memcpy(&o, &h, 2); return o;
}
__device__ __forceinline__ float bf2f(__bf16 b) {
  unsigned short h; __builtin_memcpy(&h, &b, 2);
  union { unsigned u; float f; } c; c.u = ((unsigned)h) << 16; return c.f;
}
// build a 16xbf16 WMMA fragment from two 128-bit register quads (no memory)
__device__ __forceinline__ v16bf mkfrag(uint4 a, uint4 b) {
  union { uint4 u[2]; v16bf v; } c; c.u[0] = a; c.u[1] = b; return c.v;
}

// ---------------- weight conversion ----------------
__global__ void k_cvt_bf16(const float* __restrict__ src, __bf16* __restrict__ dst, int n) {
  int i = blockIdx.x * 256 + threadIdx.x;
  if (i < n) dst[i] = f2bf(src[i]);
}
// pad w_ml2 (19x256) -> (32x256) zero rows
__global__ void k_pad_ml2(const float* __restrict__ src, __bf16* __restrict__ dst) {
  int i = blockIdx.x * 256 + threadIdx.x;   // 32*256 = 8192
  if (i < 32 * kC) {
    int row = i / kC;
    dst[i] = (row < kNC) ? f2bf(src[i]) : f2bf(0.0f);
  }
}

// ---------------- K1: dual LayerNorm + fuse ----------------
__global__ void k_ln_fuse(const float* __restrict__ low, const float* __restrict__ high,
                          const float* __restrict__ gl, const float* __restrict__ bl,
                          const float* __restrict__ gh, const float* __restrict__ bh,
                          float* __restrict__ query, __bf16* __restrict__ fused) {
  int pid = blockIdx.x * 256 + threadIdx.x;      // 0..M-1
  int b = pid >> 14, p = pid & (kHW - 1);
  const float* lo = low  + (size_t)b * kC * kHW + p;
  const float* hi = high + (size_t)b * kC * kHW + p;
  float s1 = 0.f, s2 = 0.f, t1 = 0.f, t2 = 0.f;
  for (int c = 0; c < kC; ++c) {
    float v = lo[(size_t)c * kHW]; s1 += v; s2 += v * v;
    float w = hi[(size_t)c * kHW]; t1 += w; t2 += w * w;
  }
  float mu1 = s1 / kC, r1 = rsqrtf(s2 / kC - mu1 * mu1 + 1e-5f);
  float mu2 = t1 / kC, r2 = rsqrtf(t2 / kC - mu2 * mu2 + 1e-5f);
  for (int c = 0; c < kC; ++c) {
    float q = (lo[(size_t)c * kHW] - mu1) * r1 * gl[c] + bl[c];
    float k = (hi[(size_t)c * kHW] - mu2) * r2 * gh[c] + bh[c];
    query[(size_t)b * kC * kHW + (size_t)c * kHW + p] = q;
    fused[(size_t)pid * kC + c] = f2bf((q + k) * 0.5f);
  }
}

// ---------------- WMMA bf16 GEMM, 2-stage pipeline with NAMED register sets ----------------
// C[m, col] = sum_k A[m,k] * Bt[col,k]   (Bt = weight [Cout][Cin], row-major)
// One wave computes a 16 x (16*NT) tile; blockDim = 256 (8 waves). K must be a multiple of 64.
// grid.z = groups (offsets aZOff/bZOff/cZOff per group).
// modes: 0 -> bf16 out           1 -> f32 out
//        2 -> f32 out = acc+bias+resNCHW[b,col,p]   (proj + low residual)
//        3 -> dstNCHW[b,col,p] = acc+bias+res[m*ldc+col]  (mlp2 + out1 residual)
template <int NT>
__global__ void __launch_bounds__(256)
k_gemm(const __bf16* __restrict__ A, int lda, int aZOff,
       const __bf16* __restrict__ Bt, int ldb, int bZOff,
       int K, int nTilesPerGroup, int cZOff, int ldc,
       const float* __restrict__ bias, int mode,
       float* __restrict__ outF, __bf16* __restrict__ outB,
       const float* __restrict__ res, float* __restrict__ dstNCHW) {
  int wid  = (blockIdx.x * blockDim.x + threadIdx.x) >> 5;
  int lane = threadIdx.x & 31;
  int mTile = wid / nTilesPerGroup;
  int nTile = wid - mTile * nTilesPerGroup;
  int z = blockIdx.z;
  A  += (size_t)z * aZOff;
  Bt += (size_t)z * bZOff;
  int colBase = z * cZOff + nTile * (16 * NT);
  int mBase = mTile * 16;
  int half = lane >> 4, r = lane & 15;

  v8f acc[NT];
#pragma unroll
  for (int j = 0; j < NT; ++j)
#pragma unroll
    for (int e = 0; e < 8; ++e) acc[j][e] = 0.0f;

  const __bf16* aPtr = A + (size_t)(mBase + r) * lda + half * 8;
  const __bf16* bRow[NT];
#pragma unroll
  for (int j = 0; j < NT; ++j)
    bRow[j] = Bt + (size_t)(colBase + j * 16 + r) * ldb + half * 8;

  // two named pipeline stages (constant-indexed only -> stay in VGPRs)
  uint4 a0lo, a0hi, a1lo, a1hi;
  uint4 b0lo[NT], b0hi[NT], b1lo[NT], b1hi[NT];

  // preload stage0 @ k = 0
  a0lo = *(const uint4*)(aPtr);
  a0hi = *(const uint4*)(aPtr + 16);
#pragma unroll
  for (int j = 0; j < NT; ++j) {
    b0lo[j] = *(const uint4*)(bRow[j]);
    b0hi[j] = *(const uint4*)(bRow[j] + 16);
  }

  // steady state: 64-wide K super-steps, prefetch one stage ahead
  int k0 = 0;
  for (; k0 + 64 < K; k0 += 64) {
    a1lo = *(const uint4*)(aPtr + k0 + 32);
    a1hi = *(const uint4*)(aPtr + k0 + 48);
#pragma unroll
    for (int j = 0; j < NT; ++j) {
      b1lo[j] = *(const uint4*)(bRow[j] + k0 + 32);
      b1hi[j] = *(const uint4*)(bRow[j] + k0 + 48);
    }
#pragma unroll
    for (int j = 0; j < NT; ++j)
      acc[j] = __builtin_amdgcn_wmma_f32_16x16x32_bf16(
          false, mkfrag(a0lo, a0hi), false, mkfrag(b0lo[j], b0hi[j]), (short)0, acc[j], false, false);
    a0lo = *(const uint4*)(aPtr + k0 + 64);
    a0hi = *(const uint4*)(aPtr + k0 + 80);
#pragma unroll
    for (int j = 0; j < NT; ++j) {
      b0lo[j] = *(const uint4*)(bRow[j] + k0 + 64);
      b0hi[j] = *(const uint4*)(bRow[j] + k0 + 80);
    }
#pragma unroll
    for (int j = 0; j < NT; ++j)
      acc[j] = __builtin_amdgcn_wmma_f32_16x16x32_bf16(
          false, mkfrag(a1lo, a1hi), false, mkfrag(b1lo[j], b1hi[j]), (short)0, acc[j], false, false);
  }
  // tail: last 64 of K (k0 == K-64)
  a1lo = *(const uint4*)(aPtr + k0 + 32);
  a1hi = *(const uint4*)(aPtr + k0 + 48);
#pragma unroll
  for (int j = 0; j < NT; ++j) {
    b1lo[j] = *(const uint4*)(bRow[j] + k0 + 32);
    b1hi[j] = *(const uint4*)(bRow[j] + k0 + 48);
  }
#pragma unroll
  for (int j = 0; j < NT; ++j)
    acc[j] = __builtin_amdgcn_wmma_f32_16x16x32_bf16(
        false, mkfrag(a0lo, a0hi), false, mkfrag(b0lo[j], b0hi[j]), (short)0, acc[j], false, false);
#pragma unroll
  for (int j = 0; j < NT; ++j)
    acc[j] = __builtin_amdgcn_wmma_f32_16x16x32_bf16(
        false, mkfrag(a1lo, a1hi), false, mkfrag(b1lo[j], b1hi[j]), (short)0, acc[j], false, false);

  // epilogue
#pragma unroll
  for (int j = 0; j < NT; ++j) {
    int col = colBase + j * 16 + r;
    float bv = bias ? bias[col] : 0.0f;
#pragma unroll
    for (int vr = 0; vr < 8; ++vr) {
      int m = mBase + vr + half * 8;
      float val = acc[j][vr] + bv;
      if (mode == 0) {
        outB[(size_t)m * ldc + col] = f2bf(val);
      } else if (mode == 1) {
        outF[(size_t)m * ldc + col] = val;
      } else if (mode == 2) {
        int b = m >> 14, p = m & (kHW - 1);
        val += res[((size_t)b * kC + col) * kHW + p];
        outF[(size_t)m * ldc + col] = val;
      } else {
        val += res[(size_t)m * ldc + col];
        int b = m >> 14, p = m & (kHW - 1);
        dstNCHW[((size_t)b * kC + col) * kHW + p] = val;
      }
    }
  }
}

// ---------------- K3: spatial softmax of mask over HW per (b,n) ----------------
__global__ void k_softmax_spatial(float* __restrict__ mask /*[B,HW,32]*/) {
  int n = blockIdx.x, b = blockIdx.y;
  __shared__ float red[256];
  float* base = mask + (size_t)b * kHW * 32 + n;
  float mx = -1e30f;
  for (int p = threadIdx.x; p < kHW; p += 256) mx = fmaxf(mx, base[(size_t)p * 32]);
  red[threadIdx.x] = mx; __syncthreads();
  for (int s = 128; s > 0; s >>= 1) {
    if (threadIdx.x < s) red[threadIdx.x] = fmaxf(red[threadIdx.x], red[threadIdx.x + s]);
    __syncthreads();
  }
  mx = red[0]; __syncthreads();
  float sum = 0.f;
  for (int p = threadIdx.x; p < kHW; p += 256) {
    float e = __expf(base[(size_t)p * 32] - mx);
    base[(size_t)p * 32] = e; sum += e;
  }
  red[threadIdx.x] = sum; __syncthreads();
  for (int s = 128; s > 0; s >>= 1) {
    if (threadIdx.x < s) red[threadIdx.x] += red[threadIdx.x + s];
    __syncthreads();
  }
  float inv = 1.0f / red[0];
  for (int p = threadIdx.x; p < kHW; p += 256) base[(size_t)p * 32] *= inv;
}

// ---------------- K4: cf partials (no atomics, 16 spatial chunks) ----------------
constexpr int kCHUNKS = 16;
__global__ void k_cf_partial(const float* __restrict__ mask, const float* __restrict__ xa,
                             float* __restrict__ cfp /*[16][B*NC*C]*/) {
  int n = blockIdx.x, b = blockIdx.y, z = blockIdx.z, c = threadIdx.x;
  int p0 = z * (kHW / kCHUNKS), p1 = p0 + kHW / kCHUNKS;
  const float* mb = mask + (size_t)b * kHW * 32 + n;
  const float* xb = xa + (size_t)b * kHW * kC + c;
  float acc = 0.f;
  for (int p = p0; p < p1; ++p)
    acc += mb[(size_t)p * 32] * xb[(size_t)p * kC];
  cfp[((size_t)z * kB * kNC + (size_t)b * kNC + n) * kC + c] = acc;
}
__global__ void k_cf_reduce(const float* __restrict__ cfp, const float* __restrict__ memory,
                            float* __restrict__ cf) {
  int i = blockIdx.x * 256 + threadIdx.x;        // B*NC*C = 19456
  if (i >= kB * kNC * kC) return;
  float acc = 0.f;
  for (int z = 0; z < kCHUNKS; ++z) acc += cfp[(size_t)z * kB * kNC * kC + i];
  int nc = i % (kNC * kC);                        // memory is [1,NC,C]
  cf[i] = (1.0f - kMOM) * acc + kMOM * memory[nc];
}

// ---------------- K5: kv = cf @ w_kv^T + b_kv ----------------
__global__ void k_kv(const float* __restrict__ cf, const float* __restrict__ w_kv,
                     const float* __restrict__ b_kv, float* __restrict__ kv) {
  int i = blockIdx.x * 256 + threadIdx.x;        // B*NC*2C = 38912
  if (i >= kB * kNC * 2 * kC) return;
  int d = i % (2 * kC), bn = i / (2 * kC);
  const float* c0 = cf + (size_t)bn * kC;
  const float* wr = w_kv + (size_t)d * kC;
  float acc = b_kv[d];
  for (int c = 0; c < kC; ++c) acc += c0[c] * wr[c];
  kv[i] = acc;
}

// ---------------- K6a: depthwise 3x3 on query (NCHW f32 -> [M,C] bf16) ----------------
__global__ void k_dwconv_q(const float* __restrict__ query, const float* __restrict__ w,
                           const float* __restrict__ bias, __bf16* __restrict__ out) {
  int idx = blockIdx.x * 256 + threadIdx.x;      // B*C*HW, p fastest
  int p = idx & (kHW - 1); int bc = idx >> 14;   // bc = b*C + c
  int c = bc & (kC - 1);   int b = bc >> 8;
  int x = p & (kW - 1), y = p >> 7;
  const float* in = query + (size_t)bc * kHW;
  const float* wc = w + c * 9;
  float acc = bias[c];
#pragma unroll
  for (int dy = -1; dy <= 1; ++dy) {
    int yy = y + dy; if (yy < 0 || yy >= kH) continue;
#pragma unroll
    for (int dx = -1; dx <= 1; ++dx) {
      int xx = x + dx; if (xx < 0 || xx >= kW) continue;
      acc += in[yy * kW + xx] * wc[(dy + 1) * 3 + (dx + 1)];
    }
  }
  out[((size_t)b * kHW + p) * kC + c] = f2bf(acc);
}

// ---------------- K7: 19-token cross attention (max-subtract softmax) ----------------
__global__ void k_attn(const float* __restrict__ q /*[M,C]*/, const float* __restrict__ kv,
                       __bf16* __restrict__ out /*[M,C]*/) {
  int b = blockIdx.z, h = blockIdx.y;
  int p = blockIdx.x * 256 + threadIdx.x;
  __shared__ float ks[kNC * kHD], vs[kNC * kHD];
  for (int i = threadIdx.x; i < kNC * kHD; i += 256) {
    int n = i / kHD, d = i % kHD;
    ks[i] = kv[((size_t)b * kNC + n) * 2 * kC + h * kHD + d];
    vs[i] = kv[((size_t)b * kNC + n) * 2 * kC + kC + h * kHD + d];
  }
  __syncthreads();
  const float* qp = q + ((size_t)b * kHW + p) * kC + h * kHD;
  float qr[kHD];
#pragma unroll
  for (int d = 0; d < kHD; ++d) qr[d] = qp[d];
  float sc[kNC]; float mn = 1e30f;
  for (int n = 0; n < kNC; ++n) {
    float a = 0.f;
#pragma unroll
    for (int d = 0; d < kHD; ++d) a += qr[d] * ks[n * kHD + d];
    a *= kSCALE; sc[n] = a; mn = fminf(mn, a);
  }
  // attn' = max - s; softmax(attn') == exp(mn - s)/sum
  float wgt[kNC]; float sum = 0.f;
  for (int n = 0; n < kNC; ++n) { float e = __expf(mn - sc[n]); wgt[n] = e; sum += e; }
  float inv = 1.0f / sum;
  __bf16* op = out + ((size_t)b * kHW + p) * kC + h * kHD;
#pragma unroll
  for (int d = 0; d < kHD; ++d) {
    float a = 0.f;
    for (int n = 0; n < kNC; ++n) a += wgt[n] * vs[n * kHD + d];
    op[d] = f2bf(a * inv);
  }
}

// ---------------- K8b: LayerNorm of out1 -> y bf16 (wave per pixel, wave32) ----------------
__global__ void k_ln2(const float* __restrict__ out1, const float* __restrict__ g,
                      const float* __restrict__ be, __bf16* __restrict__ y) {
  int wid  = (blockIdx.x * 256 + threadIdx.x) >> 5;  // pixel row 0..M-1
  int lane = threadIdx.x & 31;
  const float* row = out1 + (size_t)wid * kC;
  float v[8]; float s = 0.f, s2 = 0.f;
#pragma unroll
  for (int j = 0; j < 8; ++j) {
    v[j] = row[lane + j * 32]; s += v[j]; s2 += v[j] * v[j];
  }
#pragma unroll
  for (int off = 16; off > 0; off >>= 1) {
    s  += __shfl_xor(s,  off, 32);
    s2 += __shfl_xor(s2, off, 32);
  }
  float mu = s / kC;
  float r = rsqrtf(s2 / kC - mu * mu + 1e-5f);
#pragma unroll
  for (int j = 0; j < 8; ++j) {
    int c = lane + j * 32;
    y[(size_t)wid * kC + c] = f2bf((v[j] - mu) * r * g[c] + be[c]);
  }
}

// ---------------- K10: depthwise 3x3 on h1 [M,1024] bf16 + exact GELU ----------------
__global__ void k_dwconv_mlp(const __bf16* __restrict__ h1, const float* __restrict__ w,
                             const float* __restrict__ bias, __bf16* __restrict__ h2) {
  size_t idx = (size_t)blockIdx.x * 256 + threadIdx.x;  // B*HW*1024, c fastest
  int c = (int)(idx & (kC4 - 1));
  size_t bp = idx >> 10;
  int p = (int)(bp & (kHW - 1)); int b = (int)(bp >> 14);
  int x = p & (kW - 1), y = p >> 7;
  const float* wc = w + c * 9;
  float acc = bias[c];
#pragma unroll
  for (int dy = -1; dy <= 1; ++dy) {
    int yy = y + dy; if (yy < 0 || yy >= kH) continue;
#pragma unroll
    for (int dx = -1; dx <= 1; ++dx) {
      int xx = x + dx; if (xx < 0 || xx >= kW) continue;
      acc += bf2f(h1[((size_t)b * kHW + yy * kW + xx) * kC4 + c]) * wc[(dy + 1) * 3 + (dx + 1)];
    }
  }
  float ge = 0.5f * acc * (1.0f + erff(acc * 0.70710678118654752f));  // exact GELU
  h2[idx] = f2bf(ge);
}

// ---------------- host launcher ----------------
extern "C" void kernel_launch(void* const* d_in, const int* in_sizes, int n_in,
                              void* d_out, int out_size, void* d_ws, size_t ws_size,
                              hipStream_t stream) {
  (void)in_sizes; (void)n_in; (void)out_size; (void)ws_size;
  const float* low    = (const float*)d_in[0];
  const float* high   = (const float*)d_in[1];
  const float* g_low  = (const float*)d_in[2];
  const float* b_low  = (const float*)d_in[3];
  const float* g_high = (const float*)d_in[4];
  const float* b_high = (const float*)d_in[5];
  const float* g_mlp  = (const float*)d_in[6];
  const float* b_mlp  = (const float*)d_in[7];
  const float* w_q_dw = (const float*)d_in[8];
  const float* b_q_dw = (const float*)d_in[9];
  const float* w_q_pw = (const float*)d_in[10];
  const float* b_q_pw = (const float*)d_in[11];
  const float* w_ml1  = (const float*)d_in[12];
  const float* w_ml2  = (const float*)d_in[13];
  const float* w_align= (const float*)d_in[14];
  const float* w_kv   = (const float*)d_in[15];
  const float* b_kv   = (const float*)d_in[16];
  const float* memory = (const float*)d_in[17];
  const float* w_proj = (const float*)d_in[18];
  const float* b_proj = (const float*)d_in[19];
  const float* w_mlp1 = (const float*)d_in[20];
  const float* b_mlp1 = (const float*)d_in[21];
  const float* w_mlp_dw=(const float*)d_in[22];
  const float* b_mlp_dw=(const float*)d_in[23];
  const float* w_mlp2 = (const float*)d_in[24];
  const float* b_mlp2 = (const float*)d_in[25];
  float* out = (float*)d_out;

  char* ws = (char*)d_ws;
  constexpr size_t MiB = 1ull << 20;
  // lifetime-aliased layout (~366 MiB)
  float*  query  = (float*)(ws + 0);               // 64MiB  [K1..K6a]
  __bf16* fused  = (__bf16*)(ws + 64 * MiB);       // 32MiB  [K1..align]
  __bf16* m1     = (__bf16*)(ws + 96 * MiB);       // 32MiB  [ml1..ml2]
  __bf16* h1     = (__bf16*)(ws + 0);              // 128MiB [mlp1..dw]  aliases query+fused+m1
  float*  xa     = (float*)(ws + 128 * MiB);       // 64MiB  [align..cf]
  float*  qf     = (float*)(ws + 128 * MiB);       // 64MiB  [qpw..attn] aliases xa
  __bf16* qdw    = (__bf16*)(ws + 192 * MiB);      // 32MiB  [dwq..qpw]
  __bf16* attn   = (__bf16*)(ws + 224 * MiB);      // 32MiB  [attn..proj]
  __bf16* h2     = (__bf16*)(ws + 128 * MiB);      // 128MiB [dw..mlp2]  aliases xa/qf+qdw+attn
  float*  mask   = (float*)(ws + 256 * MiB);       // 8MiB   [ml2..cf]
  float*  out1   = (float*)(ws + 264 * MiB);       // 64MiB  [proj..mlp2]
  __bf16* ybuf   = (__bf16*)(ws + 328 * MiB);      // 32MiB  [ln2..mlp1]
  float*  cfp    = (float*)(ws + 360 * MiB);       // 16*B*NC*C*4 ~1.2MiB
  float*  cf     = (float*)(ws + 362 * MiB);
  float*  kvbuf  = (float*)(ws + 363 * MiB);
  __bf16* wml1b  = (__bf16*)(ws + 364 * MiB);
  __bf16* wml2b  = (__bf16*)((char*)wml1b + 32768);
  __bf16* walignb= (__bf16*)((char*)wml2b + 16384);
  __bf16* wqpwb  = (__bf16*)((char*)walignb + 32768);
  __bf16* wprojb = (__bf16*)((char*)wqpwb + 131072);
  __bf16* wmlp1b = (__bf16*)((char*)wprojb + 131072);
  __bf16* wmlp2b = (__bf16*)((char*)wmlp1b + 524288);

  // weight conversions to bf16
  k_cvt_bf16<<<(kC*64+255)/256, 256, 0, stream>>>(w_ml1,  wml1b,  kC*64);
  k_pad_ml2 <<<(32*kC+255)/256, 256, 0, stream>>>(w_ml2,  wml2b);
  k_cvt_bf16<<<(kC*64+255)/256, 256, 0, stream>>>(w_align,walignb,kC*64);
  k_cvt_bf16<<<(kC*kC+255)/256, 256, 0, stream>>>(w_q_pw, wqpwb,  kC*kC);
  k_cvt_bf16<<<(kC*kC+255)/256, 256, 0, stream>>>(w_proj, wprojb, kC*kC);
  k_cvt_bf16<<<(kC4*kC+255)/256,256, 0, stream>>>(w_mlp1, wmlp1b, kC4*kC);
  k_cvt_bf16<<<(kC*kC4+255)/256,256, 0, stream>>>(w_mlp2, wmlp2b, kC*kC4);

  // K1: LN(low), LN(high), fuse
  k_ln_fuse<<<kM/256, 256, 0, stream>>>(low, high, g_low, b_low, g_high, b_high, query, fused);

  // ml1: grouped 1x1 (4 groups of 64->64), K=64, bf16 out
  k_gemm<4><<<dim3(512,1,4), 256, 0, stream>>>(fused, kC, 64, wml1b, 64, 64*64,
      64, 1, 64, kC, nullptr, 0, nullptr, m1, nullptr, nullptr);
  // ml2: 256 -> 32(pad of 19), f32 out -> mask
  k_gemm<2><<<dim3(512,1,1), 256, 0, stream>>>(m1, kC, 0, wml2b, kC, 0,
      kC, 1, 0, 32, nullptr, 1, mask, nullptr, nullptr, nullptr);
  // align: grouped 1x1, f32 out -> xa
  k_gemm<4><<<dim3(512,1,4), 256, 0, stream>>>(fused, kC, 64, walignb, 64, 64*64,
      64, 1, 64, kC, nullptr, 1, xa, nullptr, nullptr, nullptr);

  // spatial softmax over HW per (b,n)
  k_softmax_spatial<<<dim3(kNC, kB, 1), 256, 0, stream>>>(mask);
  // class prototypes cf (+memory mix in reduce)
  k_cf_partial<<<dim3(kNC, kB, kCHUNKS), 256, 0, stream>>>(mask, xa, cfp);
  k_cf_reduce<<<(kB*kNC*kC+255)/256, 256, 0, stream>>>(cfp, memory, cf);
  // kv projection
  k_kv<<<(kB*kNC*2*kC+255)/256, 256, 0, stream>>>(cf, w_kv, b_kv, kvbuf);

  // query path: depthwise 3x3 then pointwise GEMM
  k_dwconv_q<<<(kB*kC*kHW)/256, 256, 0, stream>>>(query, w_q_dw, b_q_dw, qdw);
  k_gemm<4><<<dim3(2048,1,1), 256, 0, stream>>>(qdw, kC, 0, wqpwb, kC, 0,
      kC, 4, 0, kC, b_q_pw, 1, qf, nullptr, nullptr, nullptr);

  // attention
  k_attn<<<dim3(kHW/256, kNH, kB), 256, 0, stream>>>(qf, kvbuf, attn);

  // proj + low residual -> out1
  k_gemm<4><<<dim3(2048,1,1), 256, 0, stream>>>(attn, kC, 0, wprojb, kC, 0,
      kC, 4, 0, kC, b_proj, 2, out1, nullptr, low, nullptr);

  // LN -> y
  k_ln2<<<kM*32/256, 256, 0, stream>>>(out1, g_mlp, b_mlp, ybuf);

  // mlp1: 256 -> 1024, bf16 out h1
  k_gemm<4><<<dim3(8192,1,1), 256, 0, stream>>>(ybuf, kC, 0, wmlp1b, kC, 0,
      kC, 16, 0, kC4, b_mlp1, 0, nullptr, h1, nullptr, nullptr);

  // depthwise 3x3 + GELU -> h2
  k_dwconv_mlp<<<(size_t)kB*kHW*kC4/256, 256, 0, stream>>>(h1, w_mlp_dw, b_mlp_dw, h2);

  // mlp2: 1024 -> 256, + out1 residual, write NCHW f32 to d_out
  k_gemm<4><<<dim3(2048,1,1), 256, 0, stream>>>(h2, kC4, 0, wmlp2b, kC4, 0,
      kC4, 4, 0, kC, b_mlp2, 3, nullptr, nullptr, out1, out);
}
